// Model_53798760349839
// MI455X (gfx1250) — matrix operand order repaired and checked by reading, not compiled, not run on hardware
//
#include <hip/hip_runtime.h>

typedef __attribute__((ext_vector_type(16))) _Float16 v16h;
typedef __attribute__((ext_vector_type(8)))  _Float16 v8h;
typedef __attribute__((ext_vector_type(8)))  float    v8f;

// ---------------- layer tables (compute order) ----------------
// 0 input 75->50 | 1 h1 50->75 | 2 h2 75->75 | 3 e1 75->225 | 4 e2 225->75
// 5 e3 75->3 | 6 f1 50->150 | 7 f2 150->50 | 8 f3 50->3 | 9 d1 3->50
// 10 d2 50->150 | 11 d3 150->50 | 12 h3 50->50 | 13 h4 50->125 | 14 h5 125->50
// 15 out 50->3
static __device__ __host__ constexpr int FOFF[16] = {
    0, 6144, 12288, 21504, 46080, 70656, 73728, 83968,
    94208, 96256, 98304, 108544, 118784, 122880, 131072, 139264};

#define FRAG_ELEMS 141312   // total padded f16 weight elems (incl. bias rows)

#define IH 384
#define IW 384
#define HWPIX (IH * IW)      // 147456
#define NPIX (4 * HWPIX)     // 589824 pixels
#define MROWS 32             // pixels per wave tile (one image row segment)
#define TILES (NPIX / MROWS) // 18432
#define WAVES_PER_WG 2
#define WGS (TILES / WAVES_PER_WG) // 9216

// per-wave LDS arena widths (f16 elems per row, MROWS rows)
#define B0W 256
#define B1W 128
#define B2W 32
#define BXW 64
#define WAVE_LDS (MROWS * (B0W + B1W + B2W + BXW)) // 15360 halfs = 30720 B

// ---------------- weight fragment pack kernel ----------------
// Bias folded into weights: row k==in holds bias[n] (activations carry a
// constant 1.0 in column `in`), rows k>in and cols n>=out are exact zeros.
struct PackArgs {
    const float* W[16];
    const float* Bv[16];
    int IN[16];
    int OUT[16];
    int KP[16];
    int NP[16];
};

__global__ void pack_weights(PackArgs pa, _Float16* __restrict__ frag) {
    const int layer = blockIdx.y;
    const int idx = blockIdx.x * blockDim.x + threadIdx.x;
    const int KP = pa.KP[layer], NP = pa.NP[layer];
    const int in = pa.IN[layer], out = pa.OUT[layer];
    const int KN = KP * NP;
    if (idx < KN) {
        // fragment-major layout: frag f = nt*(KP/32)+kt, 512 f16 each,
        // inside: lane-major (32 lanes x 16 halfs), per B-matrix VGPR layout:
        //   lane<16:  N=lane,     K = kt*32 + slot + 8*(slot>=8)
        //   lane>=16: N=lane-16,  K = kt*32 + 8 + slot + 8*(slot>=8)
        const int f = idx >> 9;
        const int e = idx & 511;
        const int lane = e >> 4;
        const int slot = e & 15;
        const int v = slot >> 1, h = slot & 1;
        const int ktiles = KP >> 5;
        const int nt = f / ktiles, kt = f - nt * ktiles;
        const int n = nt * 16 + (lane & 15);
        const int k = kt * 32 + ((lane >> 4) << 3) + slot + ((slot >> 3) << 3);
        float val = 0.0f;
        if (n < out) {
            if (k < in) val = pa.W[layer][n * in + k];
            else if (k == in) val = pa.Bv[layer][n];
        }
        frag[FOFF[layer] + idx] = (_Float16)val;
    }
}

// ---------------- device helpers ----------------
__device__ __forceinline__ v16h ldpair(const _Float16* p, int gap) {
    const v8h c0 = *(const v8h*)(p);
    const v8h c1 = *(const v8h*)(p + gap);
    return __builtin_shufflevector(c0, c1, 0, 1, 2, 3, 4, 5, 6, 7, 8, 9, 10,
                                   11, 12, 13, 14, 15);
}

__device__ __forceinline__ float act_tanh(float f) {
#if __has_builtin(__builtin_amdgcn_tanhf)
    return __builtin_amdgcn_tanhf(f);
#elif __has_builtin(__builtin_amdgcn_tanh_f32)
    return __builtin_amdgcn_tanh_f32(f);
#else
    return tanhf(f);
#endif
}

// mark the constant-1.0 bias column; one row per lane, no divergence
__device__ __forceinline__ void setone(_Float16* buf, int stride, int col,
                                       int lane) {
    buf[lane * stride + col] = (_Float16)1.0f;
}

// ---------------- fused per-tile GEMM helper (M = 32 rows) ----------------
// D(32xNP) = act( A(32xKP) * B(KPxNP) [+ add] [- sub] )  (bias via 1-column)
// ACT: 0=none, 1=leaky_relu(0.01), 2=tanh
template <int KP, int NP, int ACT, bool ADD, bool SUB>
__device__ __forceinline__ void layer_gemm(
    const _Float16* __restrict__ in, int sIn,
    _Float16* __restrict__ out, int sOut,
    const _Float16* __restrict__ wfrag, int lane,
    const _Float16* __restrict__ addb, int sAdd,
    const _Float16* __restrict__ subb, int sSub) {
    constexpr int KT = KP / 32;
    constexpr int NT = NP / 16;
    constexpr int NB = (NT % 4 == 0) ? 4 : 2;  // n-tile register block
    const int nl = lane & 15;
    const int hi = (lane >> 4) & 1;
    // A-matrix 16x32 f16 layout: lane<16 -> M=lane, chunks K[0..7],K[16..23]
    //                            lane>=16 -> M=lane-16, chunks K[8..15],K[24..31]
    const _Float16* arow0 = in + nl * sIn + hi * 8;          // rows 0..15
    const _Float16* arow1 = arow0 + 16 * sIn;                // rows 16..31
    const v8f vzero = {0.f, 0.f, 0.f, 0.f, 0.f, 0.f, 0.f, 0.f};
#pragma unroll
    for (int nb = 0; nb < NT; nb += NB) {
        v8f acc0[NB], acc1[NB];
#pragma unroll
        for (int j = 0; j < NB; ++j) {
            acc0[j] = vzero;
            acc1[j] = vzero;
        }
#pragma unroll
        for (int kt = 0; kt < KT; ++kt) {
            const v16h a0 = ldpair(arow0 + kt * 32, 16);
            const v16h a1 = ldpair(arow1 + kt * 32, 16);
#pragma unroll
            for (int j = 0; j < NB; ++j) {
                const int nt = nb + j;
                const v16h b =
                    ldpair(wfrag + (nt * KT + kt) * 512 + lane * 16, 8);
                acc0[j] = __builtin_amdgcn_wmma_f32_16x16x32_f16(
                    false, a0, false, b, (short)0, acc0[j], false, false);
                acc1[j] = __builtin_amdgcn_wmma_f32_16x16x32_f16(
                    false, a1, false, b, (short)0, acc1[j], false, false);
            }
        }
        // C/D layout: VGPR v -> M = v + 8*hi (+16 for tile1), N = nt*16+(lane&15)
#pragma unroll
        for (int j = 0; j < NB; ++j) {
            const int N = (nb + j) * 16 + nl;
#pragma unroll
            for (int v = 0; v < 8; ++v) {
                {
                    float f = acc0[j][v];
                    const int M = v + hi * 8;
                    if constexpr (ADD) f += (float)addb[M * sAdd + N];
                    if constexpr (SUB) f -= (float)subb[M * sSub + N];
                    if constexpr (ACT == 1) f = (f >= 0.0f) ? f : 0.01f * f;
                    if constexpr (ACT == 2) f = act_tanh(f);
                    out[M * sOut + N] = (_Float16)f;
                }
                {
                    float f = acc1[j][v];
                    const int M = v + hi * 8 + 16;
                    if constexpr (ADD) f += (float)addb[M * sAdd + N];
                    if constexpr (SUB) f -= (float)subb[M * sSub + N];
                    if constexpr (ACT == 1) f = (f >= 0.0f) ? f : 0.01f * f;
                    if constexpr (ACT == 2) f = act_tanh(f);
                    out[M * sOut + N] = (_Float16)f;
                }
            }
        }
    }
}

__device__ __forceinline__ int refl(int i) {
    return (i < 0) ? -i : ((i > IH - 1) ? (2 * IH - 2 - i) : i);
}

// ---------------- fused MLP kernel: 1 wave per 32-pixel tile ----------------
__global__ __launch_bounds__(32 * WAVES_PER_WG) void fused_mlp(
    const float* __restrict__ x, const _Float16* __restrict__ wfrag,
    float* __restrict__ outp) {
    __shared__ _Float16 smem[WAVES_PER_WG * WAVE_LDS];
    const int wave = threadIdx.x >> 5;
    const int lane = threadIdx.x & 31;
    _Float16* b0 = smem + wave * WAVE_LDS;          // 256 wide
    _Float16* b1 = b0 + MROWS * B0W;                // 128 wide
    _Float16* b2 = b1 + MROWS * B1W;                // 32 wide
    _Float16* bx = b2 + MROWS * B2W;                // 64 wide (x1)

    const int tile = blockIdx.x * WAVES_PER_WG + wave;
    const int p0 = tile * MROWS;
    // all 32 pixels of a tile live in one image row (IW % MROWS == 0):
    // decode batch/row/column once per tile.
    const int bi = p0 / HWPIX;
    const int rem0 = p0 - bi * HWPIX;
    const int py = rem0 / IW;
    const int pxb = rem0 - py * IW;

    // ---- unfold 5x5 reflect-padded patches -> b0[32][96]
    //      (75 real + 1.0 bias column at k=75 + zeros)
    for (int kk = lane; kk < 96; kk += 32) {  // 3 outer iterations
        int c = kk / 25;
        const int r = kk - c * 25;
        const int rr = r / 5;
        const int ky = rr - 2;
        const int kx = (r - rr * 5) - 2;
        const bool isk = (kk < 75);
        if (c > 2) c = 2;  // keep address in-bounds for pad lanes
        const int iy = refl(py + ky);
        const float* __restrict__ row = x + (bi * 3 + c) * HWPIX + iy * IW;
        const _Float16 pad = (kk == 75) ? (_Float16)1.0f : (_Float16)0.0f;
        _Float16* dst = b0 + kk;
        const int ib = pxb + kx;
#pragma unroll
        for (int m = 0; m < MROWS; ++m) {
            const int ix = refl(ib + m);
            const float v = row[ix];
            dst[m * B0W] = isk ? (_Float16)v : pad;
        }
    }

#define LYR(i, KP_, NP_, ACT_, INB, SIN, OUB, SOU)                    \
    layer_gemm<KP_, NP_, ACT_, false, false>(INB, SIN, OUB, SOU,      \
                                             wfrag + FOFF[i], lane,   \
                                             nullptr, 0, nullptr, 0)

    LYR(0, 96, 64, 0, b0, B0W, bx, BXW);    // x1 = input(sw)
    setone(bx, BXW, 50, lane);
    LYR(1, 64, 96, 1, bx, BXW, b0, B0W);    // x2 = lrelu(h1(x1))
    setone(b0, B0W, 75, lane);
    LYR(2, 96, 96, 2, b0, B0W, b1, B1W);    // x3 = tanh(h2(x2))
    setone(b1, B1W, 75, lane);
    LYR(3, 96, 256, 0, b1, B1W, b0, B0W);   // t  = e1(x3)
    setone(b0, B0W, 225, lane);
    LYR(4, 256, 96, 0, b0, B0W, b1, B1W);   // t2 = e2(t)
    setone(b1, B1W, 75, lane);
    LYR(5, 96, 32, 0, b1, B1W, b2, B2W);    // enc3 = e3(t2)  (in b2)
    LYR(6, 64, 160, 0, bx, BXW, b0, B0W);   // u  = f1(x1)
    setone(b0, B0W, 150, lane);
    LYR(7, 160, 64, 0, b0, B0W, b1, B1W);   // u2 = f2(u)
    setone(b1, B1W, 50, lane);
    // x4 = lrelu(f3(u2) + enc3)  -> b0
    layer_gemm<64, 32, 1, true, false>(b1, B1W, b0, B0W, wfrag + FOFF[8],
                                       lane, b2, B2W, nullptr, 0);
    setone(b0, B0W, 3, lane);
    LYR(9, 32, 64, 0, b0, B0W, b1, B1W);    // d1(x4)
    setone(b1, B1W, 50, lane);
    LYR(10, 64, 160, 0, b1, B1W, b0, B0W);  // d2
    setone(b0, B0W, 150, lane);
    LYR(11, 160, 64, 0, b0, B0W, b1, B1W);  // d3
    setone(b1, B1W, 50, lane);
    LYR(12, 64, 64, 0, b1, B1W, b0, B0W);   // h3 -> x4_dec
    setone(b0, B0W, 50, lane);
    LYR(13, 64, 128, 2, b0, B0W, b1, B1W);  // x5 = tanh(h4(x4_dec))
    setone(b1, B1W, 125, lane);
    // x6 = h5(x5) - x1  -> b0
    layer_gemm<128, 64, 0, false, true>(b1, B1W, b0, B0W, wfrag + FOFF[14],
                                        lane, nullptr, 0, bx, BXW);
    setone(b0, B0W, 50, lane);
    LYR(15, 64, 32, 1, b0, B0W, b1, B1W);   // x7 = lrelu(out(x6))
#undef LYR

    // ---- store x7[:, 0:3] to (b, 3, h, w): 3 coalesced b32 stores
    const int obase = bi * 3 * HWPIX + rem0 + lane;
#pragma unroll
    for (int c = 0; c < 3; ++c) {
        outp[obase + c * HWPIX] = (float)b1[lane * B1W + c];
    }
}

// ---------------- host launcher ----------------
extern "C" void kernel_launch(void* const* d_in, const int* in_sizes, int n_in,
                              void* d_out, int out_size, void* d_ws,
                              size_t ws_size, hipStream_t stream) {
    (void)in_sizes; (void)n_in; (void)out_size; (void)ws_size;

    const float* x = (const float*)d_in[0];
    _Float16* frag = (_Float16*)d_ws;

    // compute-order -> input-index mapping (W at widx, b at widx+1)
    static const int widx[16] = {1, 3, 5, 15, 17, 19, 21, 23,
                                 25, 27, 29, 31, 7, 9, 11, 13};
    static const int INs[16] = {75, 50, 75, 75, 225, 75, 50, 150,
                                50, 3, 50, 150, 50, 50, 125, 50};
    static const int OUTs[16] = {50, 75, 75, 225, 75, 3, 150, 50,
                                 3, 50, 150, 50, 50, 125, 50, 3};
    static const int KPs[16] = {96, 64, 96, 96, 256, 96, 64, 160,
                                64, 32, 64, 160, 64, 64, 128, 64};
    static const int NPs[16] = {64, 96, 96, 256, 96, 32, 160, 64,
                                32, 64, 160, 64, 64, 128, 64, 32};

    PackArgs pa;
    for (int i = 0; i < 16; ++i) {
        pa.W[i] = (const float*)d_in[widx[i]];
        pa.Bv[i] = (const float*)d_in[widx[i] + 1];
        pa.IN[i] = INs[i];
        pa.OUT[i] = OUTs[i];
        pa.KP[i] = KPs[i];
        pa.NP[i] = NPs[i];
    }

    // max KP*NP = 256*96 = 24576 -> 96 blocks of 256 threads per layer
    pack_weights<<<dim3(96, 16), 256, 0, stream>>>(pa, frag);

    fused_mlp<<<WGS, 32 * WAVES_PER_WG, 0, stream>>>(x, frag, (float*)d_out);
}
